// BaseTimeAttention_12987981103150
// MI455X (gfx1250) — matrix operand order repaired and checked by reading, not compiled, not run on hardware
//
#include <hip/hip_runtime.h>
#include <hip/hip_bf16.h>
#include <stdint.h>

// ---------------------------------------------------------------------------
// Types
// ---------------------------------------------------------------------------
typedef __bf16 bf16_t;
typedef bf16_t       v16bf __attribute__((ext_vector_type(16)));
typedef float        v8f   __attribute__((ext_vector_type(8)));
typedef float        v4f   __attribute__((ext_vector_type(4)));
typedef unsigned int v4u   __attribute__((ext_vector_type(4)));
typedef int          v4i   __attribute__((ext_vector_type(4)));
typedef int          v8i   __attribute__((ext_vector_type(8)));

union FragU {
  v16bf v;
  v4u   u[2];
};

// Problem constants (fixed by the reference).
constexpr int BATCH = 2;
constexpr int SEQ   = 2048;
constexpr int HID   = 2048;
constexpr int HEADS = 16;
constexpr int HDIM  = 128;
constexpr int MROWS = BATCH * SEQ;   // 4096

// ---------------------------------------------------------------------------
// Feature probes (evaluated separately for host/device passes).
// ---------------------------------------------------------------------------
#ifndef __has_builtin
#define __has_builtin(x) 0
#endif
#if __has_builtin(__builtin_amdgcn_tensor_load_to_lds)
#define HAVE_TDM 1
#else
#define HAVE_TDM 0
#endif
#if __has_builtin(__builtin_amdgcn_global_load_async_to_lds_b128)
#define HAVE_ASYNC 1
#else
#define HAVE_ASYNC 0
#endif

__device__ __forceinline__ unsigned lds_off(const void* p) {
  // Generic pointers to LDS carry the byte offset in the low 32 bits.
  return (unsigned)(uintptr_t)p;
}

__device__ __forceinline__ void wait_tensorcnt0() {
#if __has_builtin(__builtin_amdgcn_s_wait_tensorcnt)
  __builtin_amdgcn_s_wait_tensorcnt(0);
#else
  asm volatile("s_wait_tensorcnt 0x0" ::: "memory");
#endif
}

#if HAVE_TDM
// 2D tile DMA: global (row-major, element=2B) -> LDS, with LDS row padding.
// Bit packing per CDNA5 ISA D# Group0/Group1 tables.
__device__ __forceinline__ void tdm_load_2d(unsigned lds_addr, const void* gsrc,
                                            unsigned tensor_d0, unsigned tensor_d1,
                                            unsigned row_stride,   // elements
                                            unsigned tile_d0, unsigned tile_d1,
                                            unsigned pad_int, unsigned pad_amt) {
  const unsigned long long ga = (unsigned long long)(uintptr_t)gsrc;
  const v4u g0 = {
      1u,                                                 // count=1 (user D#)
      lds_addr,                                           // LDS byte address
      (unsigned)(ga & 0xFFFFFFFFu),                       // global addr lo
      (unsigned)((ga >> 32) & 0x01FFFFFFu) | (2u << 30)   // addr hi | type=2
  };
  const unsigned w0 = (1u << 16)            // data_size = 2 bytes
                    | (1u << 20)            // pad_enable
                    | (pad_int << 22)       // pad_interval code
                    | (pad_amt << 25);      // pad_amount code
  const v8i g1 = {
      (int)w0,
      (int)((tensor_d0 & 0xFFFFu) << 16),
      (int)(((tensor_d0 >> 16) & 0xFFFFu) | ((tensor_d1 & 0xFFFFu) << 16)),
      (int)(((tensor_d1 >> 16) & 0xFFFFu) | ((tile_d0 & 0xFFFFu) << 16)),
      (int)(tile_d1 & 0xFFFFu),
      (int)row_stride,                       // dim0 stride low 32
      0, 0
  };
  const v4i gz = {0, 0, 0, 0};
#if __clang_major__ >= 23
  const v8i gz8 = {0, 0, 0, 0, 0, 0, 0, 0};
  __builtin_amdgcn_tensor_load_to_lds(g0, g1, gz, gz, gz8, 0);
#else
  __builtin_amdgcn_tensor_load_to_lds(g0, g1, gz, gz, 0);
#endif
}
#endif  // HAVE_TDM

#if HAVE_ASYNC
// Param 1: addrspace(1) v4i* (global src); param 2: addrspace(3) v4i* (LDS dst)
typedef __attribute__((address_space(1))) v4i gv4i_t;
typedef __attribute__((address_space(3))) v4i lv4i_t;
__device__ __forceinline__ void async_cp16(bf16_t* lds, const bf16_t* g) {
  __builtin_amdgcn_global_load_async_to_lds_b128(
      (gv4i_t*)(uintptr_t)g, (lv4i_t*)(unsigned)(uintptr_t)lds, 0, 0);
}
__device__ __forceinline__ void wait_async0() {
#if __has_builtin(__builtin_amdgcn_s_wait_asynccnt)
  __builtin_amdgcn_s_wait_asynccnt(0);
#else
  asm volatile("s_wait_asynccnt 0x0" ::: "memory");
#endif
}
__device__ __forceinline__ void wait_async4() {
#if __has_builtin(__builtin_amdgcn_s_wait_asynccnt)
  __builtin_amdgcn_s_wait_asynccnt(4);
#else
  asm volatile("s_wait_asynccnt 0x4" ::: "memory");
#endif
}
#endif  // HAVE_ASYNC

// ---------------------------------------------------------------------------
// WMMA fragment helpers (CDNA5 16x16x32 bf16 layouts, wave32)
// ---------------------------------------------------------------------------
__device__ __forceinline__ v16bf load_fragA(const bf16_t* src, int ld) {
  const int lane = threadIdx.x & 31;
  const int row  = lane & 15;
  const int koff = (lane >> 4) << 3;   // 0 or 8
  FragU f;
  f.u[0] = *reinterpret_cast<const v4u*>(src + row * ld + koff);
  f.u[1] = *reinterpret_cast<const v4u*>(src + row * ld + koff + 16);
  return f.v;
}

__device__ __forceinline__ v16bf load_fragB(const bf16_t* src, int ld) {
  const int lane = threadIdx.x & 31;
  const int col  = lane & 15;
  const int kb   = (lane >> 4) << 3;   // 0 or 8
  FragU f;
  f.u[0] = *reinterpret_cast<const v4u*>(src + col * ld + kb);
  f.u[1] = *reinterpret_cast<const v4u*>(src + col * ld + kb + 16);
  return f.v;
}

__device__ __forceinline__ v8f wmma_bf16(v16bf a, v16bf b, v8f c) {
  return __builtin_amdgcn_wmma_f32_16x16x32_bf16(false, a, false, b,
                                                 (short)0, c, false, false);
}

// Convert 32 fp32 register elements -> 32 bf16 into LDS.
__device__ __forceinline__ void regs_to_lds_bf16(const v4f* r, bf16_t* s) {
  alignas(16) bf16_t tmp[32];
#pragma unroll
  for (int q = 0; q < 8; ++q) {
    tmp[q * 4 + 0] = (bf16_t)r[q].x;
    tmp[q * 4 + 1] = (bf16_t)r[q].y;
    tmp[q * 4 + 2] = (bf16_t)r[q].z;
    tmp[q * 4 + 3] = (bf16_t)r[q].w;
  }
#pragma unroll
  for (int q = 0; q < 4; ++q)
    reinterpret_cast<v4u*>(s)[q] = reinterpret_cast<const v4u*>(tmp)[q];
}

// ---------------------------------------------------------------------------
// GEMM tiling: block = 128x128 output, K-step 64, 256 threads = 8 waves (4x2),
// each wave computes 32x64 via 2x4 C-fragments.
// ---------------------------------------------------------------------------
constexpr int TM  = 128;
constexpr int TN  = 128;
constexpr int TK  = 64;
constexpr int LDK = TK + 8;   // 72 elements = 144B rows (16B aligned)

// Out(bf16)[M,N] = A(f32)[M,K] * W(f32)[N,K]^T     (Q/K/V projections)
__global__ __launch_bounds__(256) void gemm_proj(const float* __restrict__ A,
                                                 const float* __restrict__ W,
                                                 bf16_t* __restrict__ Out) {
  __shared__ alignas(16) bf16_t As[TM * LDK];
  __shared__ alignas(16) bf16_t Bs[TN * LDK];
  const int tid  = threadIdx.x;
  const int wave = tid >> 5;
  const int lane = tid & 31;
  const int m0 = blockIdx.y * TM;
  const int n0 = blockIdx.x * TN;
  const int wm = (wave >> 1) * 32;
  const int wn = (wave & 1) * 64;

  const v8f vzero = {0.f, 0.f, 0.f, 0.f, 0.f, 0.f, 0.f, 0.f};
  v8f acc[2][4];
#pragma unroll
  for (int i = 0; i < 2; ++i)
#pragma unroll
    for (int j = 0; j < 4; ++j) acc[i][j] = vzero;

  const int lrow = tid >> 1;          // 0..127 tile row
  const int lcol = (tid & 1) * 32;    // 0 or 32
  const float* pa = A + (size_t)(m0 + lrow) * HID + lcol;
  const float* pw = W + (size_t)(n0 + lrow) * HID + lcol;

  v4f ar[8], wr[8];
#pragma unroll
  for (int q = 0; q < 8; ++q) {
    ar[q] = reinterpret_cast<const v4f*>(pa)[q];
    wr[q] = reinterpret_cast<const v4f*>(pw)[q];
  }

  for (int k0 = 0; k0 < HID; k0 += TK) {
    regs_to_lds_bf16(ar, As + lrow * LDK + lcol);
    regs_to_lds_bf16(wr, Bs + lrow * LDK + lcol);
    __syncthreads();

    // Issue next tile's global loads now; they complete under the WMMAs.
    const int kn = (k0 + TK < HID) ? (k0 + TK) : 0;
#pragma unroll
    for (int q = 0; q < 8; ++q) {
      ar[q] = reinterpret_cast<const v4f*>(pa + kn)[q];
      wr[q] = reinterpret_cast<const v4f*>(pw + kn)[q];
    }
    if (k0 + 2 * TK < HID) {
      __builtin_prefetch(pa + k0 + 2 * TK, 0, 1);
      __builtin_prefetch(pw + k0 + 2 * TK, 0, 1);
    }

#pragma unroll
    for (int kk = 0; kk < TK; kk += 32) {
      v16bf af[2], bfr[4];
#pragma unroll
      for (int i = 0; i < 2; ++i)
        af[i] = load_fragA(As + (wm + i * 16) * LDK + kk, LDK);
#pragma unroll
      for (int j = 0; j < 4; ++j)
        bfr[j] = load_fragB(Bs + (wn + j * 16) * LDK + kk, LDK);
#pragma unroll
      for (int i = 0; i < 2; ++i)
#pragma unroll
        for (int j = 0; j < 4; ++j)
          acc[i][j] = wmma_bf16(af[i], bfr[j], acc[i][j]);
    }
    __syncthreads();
  }

  const int rb = m0 + wm + ((lane >> 4) << 3);
  const int cb = n0 + wn + (lane & 15);
#pragma unroll
  for (int i = 0; i < 2; ++i)
#pragma unroll
    for (int j = 0; j < 4; ++j)
#pragma unroll
      for (int r = 0; r < 8; ++r)
        Out[(size_t)(rb + i * 16 + r) * HID + cb + j * 16] =
            (bf16_t)acc[i][j][r];
}

// Out(f32)[M,N] = A(bf16)[M,K] * W(f32)[N,K]^T      (output projection)
// A tile (already bf16) is staged global->LDS with double-buffered async
// copies (ASYNCcnt pipeline); W tile goes through the fp32->bf16 register
// pipeline as in gemm_proj.
__global__ __launch_bounds__(256) void gemm_out(const bf16_t* __restrict__ A,
                                                const float* __restrict__ W,
                                                float* __restrict__ Out) {
  __shared__ alignas(16) bf16_t As[2][TM * LDK];
  __shared__ alignas(16) bf16_t Bs[TN * LDK];
  const int tid  = threadIdx.x;
  const int wave = tid >> 5;
  const int lane = tid & 31;
  const int m0 = blockIdx.y * TM;
  const int n0 = blockIdx.x * TN;
  const int wm = (wave >> 1) * 32;
  const int wn = (wave & 1) * 64;

  const v8f vzero = {0.f, 0.f, 0.f, 0.f, 0.f, 0.f, 0.f, 0.f};
  v8f acc[2][4];
#pragma unroll
  for (int i = 0; i < 2; ++i)
#pragma unroll
    for (int j = 0; j < 4; ++j) acc[i][j] = vzero;

  const int lrow = tid >> 1;
  const int lcol = (tid & 1) * 32;
  const float* pw = W + (size_t)(n0 + lrow) * HID + lcol;

  // Prime buffer 0 with the k0=0 A tile (4 x 16B chunks per thread).
#if HAVE_ASYNC
#pragma unroll
  for (int it = 0; it < 4; ++it) {
    const int c = it * 256 + tid;
    const int arow = c >> 3, acol = (c & 7) * 8;
    async_cp16(&As[0][arow * LDK + acol],
               A + (size_t)(m0 + arow) * HID + acol);
  }
#else
#pragma unroll
  for (int it = 0; it < 4; ++it) {
    const int c = it * 256 + tid;
    const int arow = c >> 3, acol = (c & 7) * 8;
    *reinterpret_cast<v4u*>(&As[0][arow * LDK + acol]) =
        *reinterpret_cast<const v4u*>(A + (size_t)(m0 + arow) * HID + acol);
  }
#endif

  v4f wr[8];
#pragma unroll
  for (int q = 0; q < 8; ++q) wr[q] = reinterpret_cast<const v4f*>(pw)[q];

  for (int k0 = 0; k0 < HID; k0 += TK) {
    const int buf = (k0 / TK) & 1;
    regs_to_lds_bf16(wr, Bs + lrow * LDK + lcol);

    // Issue next A tile into the other buffer (wraps on last iteration so the
    // ASYNCcnt<=4 wait below always proves the current tile completed).
    const int kn = (k0 + TK < HID) ? (k0 + TK) : 0;
#if HAVE_ASYNC
#pragma unroll
    for (int it = 0; it < 4; ++it) {
      const int c = it * 256 + tid;
      const int arow = c >> 3, acol = (c & 7) * 8;
      async_cp16(&As[buf ^ 1][arow * LDK + acol],
                 A + (size_t)(m0 + arow) * HID + kn + acol);
    }
#else
#pragma unroll
    for (int it = 0; it < 4; ++it) {
      const int c = it * 256 + tid;
      const int arow = c >> 3, acol = (c & 7) * 8;
      *reinterpret_cast<v4u*>(&As[buf ^ 1][arow * LDK + acol]) =
          *reinterpret_cast<const v4u*>(A + (size_t)(m0 + arow) * HID + kn + acol);
    }
#endif
    // W regs for the next iteration; loads retire under the WMMAs.
#pragma unroll
    for (int q = 0; q < 8; ++q)
      wr[q] = reinterpret_cast<const v4f*>(pw + kn)[q];
    if (k0 + 2 * TK < HID) __builtin_prefetch(pw + k0 + 2 * TK, 0, 1);

#if HAVE_ASYNC
    wait_async4();   // 4 newest outstanding -> current tile's 4 are done
#endif
    __syncthreads();

#pragma unroll
    for (int kk = 0; kk < TK; kk += 32) {
      v16bf af[2], bfr[4];
#pragma unroll
      for (int i = 0; i < 2; ++i)
        af[i] = load_fragA(&As[buf][(wm + i * 16) * LDK + kk], LDK);
#pragma unroll
      for (int j = 0; j < 4; ++j)
        bfr[j] = load_fragB(Bs + (wn + j * 16) * LDK + kk, LDK);
#pragma unroll
      for (int i = 0; i < 2; ++i)
#pragma unroll
        for (int j = 0; j < 4; ++j)
          acc[i][j] = wmma_bf16(af[i], bfr[j], acc[i][j]);
    }
    __syncthreads();
  }

  const int rb = m0 + wm + ((lane >> 4) << 3);
  const int cb = n0 + wn + (lane & 15);
#pragma unroll
  for (int i = 0; i < 2; ++i)
#pragma unroll
    for (int j = 0; j < 4; ++j)
#pragma unroll
      for (int r = 0; r < 8; ++r)
        Out[(size_t)(rb + i * 16 + r) * HID + cb + j * 16] = acc[i][j][r];
}

// ---------------------------------------------------------------------------
// RoPE (in place on Q,K bf16) + transpose V into Vt[b][h][d][s] (bf16)
// ---------------------------------------------------------------------------
__global__ __launch_bounds__(128) void rope_vt(bf16_t* __restrict__ Q,
                                               bf16_t* __restrict__ K,
                                               const bf16_t* __restrict__ V,
                                               bf16_t* __restrict__ Vt) {
  const int d = threadIdx.x;   // 0..127
  const int s = blockIdx.x;
  const int h = blockIdx.y;
  const int b = blockIdx.z;
  const size_t row = (size_t)(b * SEQ + s) * HID + h * HDIM;

  Vt[((size_t)((b * HEADS + h) * HDIM + d)) * SEQ + s] = V[row + d];

  if (d < 64) {
    const float inv_freq = exp2f(-((float)(2 * d) * (1.0f / 128.0f)) *
                                 13.287712379549449f);     // log2(10000)
    const float ang = (float)s * inv_freq;
    const float cs = __cosf(ang);
    const float sn = __sinf(ang);

    float q1 = (float)Q[row + d];
    float q2 = (float)Q[row + d + 64];
    Q[row + d]      = (bf16_t)(q1 * cs - q2 * sn);
    Q[row + d + 64] = (bf16_t)(q1 * sn + q2 * cs);

    float k1 = (float)K[row + d];
    float k2 = (float)K[row + d + 64];
    K[row + d]      = (bf16_t)(k1 * cs - k2 * sn);
    K[row + d + 64] = (bf16_t)(k1 * sn + k2 * cs);
  }
}

// ---------------------------------------------------------------------------
// Flash attention: block = 4 waves sharing one K/V tile staged into LDS via
// TDM (preferred) / async global->LDS / manual copies. Each wave owns a
// 16-row Q stripe. Softmax row-reductions via shfl_xor in 16-lane groups.
// ---------------------------------------------------------------------------
constexpr int KLD = 136;  // K tile LDS row stride (64 + 4 DWORD pad)
constexpr int VLD = 40;   // V tile LDS row stride (16 + 4 DWORD pad)

__device__ __forceinline__ float redmax16(float v) {
#pragma unroll
  for (int off = 1; off < 16; off <<= 1)
    v = fmaxf(v, __shfl_xor(v, off, 32));
  return v;
}
__device__ __forceinline__ float redsum16(float v) {
#pragma unroll
  for (int off = 1; off < 16; off <<= 1) v += __shfl_xor(v, off, 32);
  return v;
}

__global__ __launch_bounds__(128) void attn_fa(const bf16_t* __restrict__ Q,
                                               const bf16_t* __restrict__ K,
                                               const bf16_t* __restrict__ Vt,
                                               bf16_t* __restrict__ O) {
  const int wave = threadIdx.x >> 5;
  const int lane = threadIdx.x & 31;
  const int h = blockIdx.y;
  const int b = blockIdx.z;
  const int q0 = blockIdx.x * 64 + wave * 16;

  const float sc = 0.08838834764831845f * 1.4426950408889634f;  // 1/sqrt(128)*log2e

  __shared__ alignas(16) bf16_t Ks[32 * KLD];    // 8704 B
  __shared__ alignas(16) bf16_t Vs[128 * VLD];   // 10240 B
  __shared__ alignas(16) bf16_t Ps[4][16 * 40];  // per-wave P scratch

  const bf16_t* qbase = Q + (size_t)(b * SEQ + q0) * HID + h * HDIM;
  const bf16_t* kbase = K + (size_t)(b * SEQ) * HID + h * HDIM;
  const bf16_t* vbase = Vt + (size_t)((b * HEADS + h) * HDIM) * SEQ;

  v16bf qf[4];
#pragma unroll
  for (int c = 0; c < 4; ++c) qf[c] = load_fragA(qbase + c * 32, HID);

  const v8f vzero = {0.f, 0.f, 0.f, 0.f, 0.f, 0.f, 0.f, 0.f};
  v8f o[8];
#pragma unroll
  for (int nt = 0; nt < 8; ++nt) o[nt] = vzero;
  float mrow[8], lrow[8];
#pragma unroll
  for (int i = 0; i < 8; ++i) { mrow[i] = -3.0e38f; lrow[i] = 0.f; }

  for (int s0 = 0; s0 < SEQ; s0 += 32) {
    // ---- stage K (32 rows x 128) and V^T (128 rows x 32) tiles into LDS
#if HAVE_TDM
    if (wave == 0) {
      // K: tensor 2048x2048(stride HID), tile 128x32, pad 4 DW per 64 DW
      tdm_load_2d(lds_off(Ks), kbase + (size_t)s0 * HID,
                  HID, SEQ, HID, HDIM, 32, 5, 3);
      wait_tensorcnt0();
    } else if (wave == 1) {
      // V^T: tensor 2048x128(stride SEQ), tile 32x128, pad 4 DW per 16 DW
      tdm_load_2d(lds_off(Vs), vbase + s0,
                  SEQ, HDIM, SEQ, 32, HDIM, 3, 3);
      wait_tensorcnt0();
    }
#elif HAVE_ASYNC
#pragma unroll
    for (int it = 0; it < 4; ++it) {
      const int c = it * 128 + threadIdx.x;
      const int kr = c >> 4, kc = c & 15;
      async_cp16(Ks + kr * KLD + kc * 8,
                 kbase + (size_t)(s0 + kr) * HID + kc * 8);
      const int vr = c >> 2, vc = c & 3;
      async_cp16(Vs + vr * VLD + vc * 8,
                 vbase + (size_t)vr * SEQ + s0 + vc * 8);
    }
    wait_async0();
#else
#pragma unroll
    for (int it = 0; it < 4; ++it) {
      const int c = it * 128 + threadIdx.x;
      const int kr = c >> 4, kc = c & 15;
      *reinterpret_cast<v4u*>(Ks + kr * KLD + kc * 8) =
          *reinterpret_cast<const v4u*>(kbase + (size_t)(s0 + kr) * HID + kc * 8);
      const int vr = c >> 2, vc = c & 3;
      *reinterpret_cast<v4u*>(Vs + vr * VLD + vc * 8) =
          *reinterpret_cast<const v4u*>(vbase + (size_t)vr * SEQ + s0 + vc * 8);
    }
#endif
    __syncthreads();

    // ---- scores = Q[16x128] . K^T[128x32] -> two 16x16 C fragments
    v8f sf[2] = {vzero, vzero};
#pragma unroll
    for (int nt = 0; nt < 2; ++nt)
#pragma unroll
      for (int c = 0; c < 4; ++c) {
        v16bf kf = load_fragB(Ks + (nt * 16) * KLD + c * 32, KLD);
        sf[nt] = wmma_bf16(qf[c], kf, sf[nt]);
      }

    // ---- online softmax
    float mnew[8];
#pragma unroll
    for (int i = 0; i < 8; ++i) {
      float r = fmaxf(sf[0][i], sf[1][i]);
      r = redmax16(r);
      mnew[i] = fmaxf(mrow[i], r);
    }
    float p0[8], p1[8], alpha[8];
#pragma unroll
    for (int i = 0; i < 8; ++i) {
      alpha[i] = exp2f((mrow[i] - mnew[i]) * sc);
      p0[i] = exp2f((sf[0][i] - mnew[i]) * sc);
      p1[i] = exp2f((sf[1][i] - mnew[i]) * sc);
      lrow[i] = lrow[i] * alpha[i] + redsum16(p0[i] + p1[i]);
      mrow[i] = mnew[i];
    }
#pragma unroll
    for (int nt = 0; nt < 8; ++nt)
#pragma unroll
      for (int i = 0; i < 8; ++i) o[nt][i] *= alpha[i];

    // ---- re-layout P (C-frag -> A-frag) through per-wave LDS scratch
    bf16_t* pr = &Ps[wave][0];
    const int rsel = (lane >> 4) << 3;
#pragma unroll
    for (int i = 0; i < 8; ++i) {
      pr[(rsel + i) * 40 + (lane & 15)]      = (bf16_t)p0[i];
      pr[(rsel + i) * 40 + 16 + (lane & 15)] = (bf16_t)p1[i];
    }
    v16bf pf = load_fragA(pr, 40);

    // ---- O += P[16x32] . V[32x128]  (V^T tile: B operand K-contiguous)
#pragma unroll
    for (int nt = 0; nt < 8; ++nt) {
      v16bf vf = load_fragB(Vs + (nt * 16) * VLD, VLD);
      o[nt] = wmma_bf16(pf, vf, o[nt]);
    }
    __syncthreads();   // protect K/V tiles before next stage overwrites
  }

  // ---- normalize and write out (row layout [b*S+q, h*128+d])
  bf16_t* obase = O + (size_t)(b * SEQ + q0) * HID + h * HDIM;
  const int rsel = (lane >> 4) << 3;
  float inv[8];
#pragma unroll
  for (int i = 0; i < 8; ++i) inv[i] = 1.0f / lrow[i];
#pragma unroll
  for (int nt = 0; nt < 8; ++nt)
#pragma unroll
    for (int i = 0; i < 8; ++i)
      obase[(size_t)(rsel + i) * HID + nt * 16 + (lane & 15)] =
          (bf16_t)(o[nt][i] * inv[i]);
}

// ---------------------------------------------------------------------------
// Launch
// ---------------------------------------------------------------------------
extern "C" void kernel_launch(void* const* d_in, const int* in_sizes, int n_in,
                              void* d_out, int out_size, void* d_ws,
                              size_t ws_size, hipStream_t stream) {
  const float* x  = (const float*)d_in[0];
  const float* Wq = (const float*)d_in[1];
  const float* Wk = (const float*)d_in[2];
  const float* Wv = (const float*)d_in[3];
  const float* Wo = (const float*)d_in[4];

  const size_t NE = (size_t)MROWS * HID;   // 8,388,608 elements
  bf16_t* Qb = (bf16_t*)d_ws;
  bf16_t* Kb = Qb + NE;
  bf16_t* Vb = Kb + NE;
  bf16_t* Vt = Vb + NE;
  bf16_t* At = Vt + NE;

  dim3 gg(HID / TN, MROWS / TM);           // (16, 32)
  gemm_proj<<<gg, 256, 0, stream>>>(x, Wq, Qb);
  gemm_proj<<<gg, 256, 0, stream>>>(x, Wk, Kb);
  gemm_proj<<<gg, 256, 0, stream>>>(x, Wv, Vb);

  rope_vt<<<dim3(SEQ, HEADS, BATCH), 128, 0, stream>>>(Qb, Kb, Vb, Vt);

  attn_fa<<<dim3(SEQ / 64, HEADS, BATCH), 128, 0, stream>>>(Qb, Kb, Vt, At);

  gemm_out<<<gg, 256, 0, stream>>>(At, Wo, (float*)d_out);
}
